// CrossCondGPTBase_18665927868870
// MI455X (gfx1250) — matrix operand order repaired and checked by reading, not compiled, not run on hardware
//
#include <hip/hip_runtime.h>
#include <math.h>

// ---------------- problem constants ----------------
#define LAYERS 12
#define DMODEL 768
#define NHEAD  12
#define HDIM   64
#define SEGLEN 512
#define SEQ    1536           // 3 * SEGLEN
#define BATCH  8
#define NMEAS  438
#define DFF    3072
#define MTOK   (BATCH * SEQ)  // 12288 token rows

typedef __attribute__((ext_vector_type(16))) __bf16 v16bf;
typedef __attribute__((ext_vector_type(8)))  __bf16 v8bf;
typedef __attribute__((ext_vector_type(2)))  __bf16 v2bf;
typedef __attribute__((ext_vector_type(8)))  float  v8f;

union V16U { v16bf v; v8bf h[2]; };

// native cvt: backend emits v_cvt_{pk_}bf16_f32 (RNE) instead of bit-twiddling
static __device__ __forceinline__ __bf16 f2bf(float f) { return (__bf16)f; }

static __device__ __forceinline__ v16bf pack16(v8bf lo, v8bf hi) {
  V16U u; u.h[0] = lo; u.h[1] = hi; return u.v;
}

// low 32 bits of a generic pointer to __shared__ = LDS byte offset
// (LDS aperture occupies addr[63:32] on gfx1250)
static __device__ __forceinline__ unsigned lds_off(const void* p) {
  return (unsigned)(size_t)p;
}

// ---------------------------------------------------------------------------
// Embedding: x[b, t, :] = {cond@cond_w+cond_b | tok_emb_up | tok_emb_down} + pos
// ---------------------------------------------------------------------------
__global__ __launch_bounds__(256) void embed_kernel(
    const int* __restrict__ idx_up, const int* __restrict__ idx_down,
    const float* __restrict__ cond, const float* __restrict__ teu,
    const float* __restrict__ ted, const float* __restrict__ pos,
    const float* __restrict__ cw, const float* __restrict__ cb,
    float* __restrict__ x) {
  const int tok = blockIdx.x;
  const int b   = tok / SEQ;
  const int tg  = tok % SEQ;
  const int seg = tg >> 9;           // SEGLEN = 512
  const int tl  = tg & (SEGLEN - 1);
  const int tid = threadIdx.x;

  __shared__ float crow[NMEAS];
  if (seg == 0) {
    for (int i = tid; i < NMEAS; i += 256)
      crow[i] = cond[(size_t)(b * SEGLEN + tl) * NMEAS + i];
  }
  __syncthreads();

  int row_idx = 0;
  if (seg == 1) row_idx = idx_up[b * SEGLEN + tl];
  else if (seg == 2) row_idx = idx_down[b * SEGLEN + tl];

  for (int d = tid; d < DMODEL; d += 256) {
    float acc;
    if (seg == 0) {
      acc = cb[d];
      for (int k = 0; k < NMEAS; ++k) acc += crow[k] * cw[(size_t)k * DMODEL + d];
    } else if (seg == 1) {
      acc = teu[(size_t)row_idx * DMODEL + d];
    } else {
      acc = ted[(size_t)row_idx * DMODEL + d];
    }
    x[(size_t)tok * DMODEL + d] = acc + pos[(size_t)tg * DMODEL + d];
  }
}

// ---------------------------------------------------------------------------
// LayerNorm (f32 in) -> bf16 out (GEMM operand)
// ---------------------------------------------------------------------------
__global__ __launch_bounds__(256) void ln_kernel(
    const float* __restrict__ x, const float* __restrict__ w,
    const float* __restrict__ b, __bf16* __restrict__ out) {
  const int tok = blockIdx.x;
  const int tid = threadIdx.x;
  __shared__ float r1[256];
  __shared__ float r2[256];
  const float* xr = x + (size_t)tok * DMODEL;
  float v0 = xr[tid], v1 = xr[tid + 256], v2 = xr[tid + 512];
  r1[tid] = v0 + v1 + v2;
  r2[tid] = v0 * v0 + v1 * v1 + v2 * v2;
  __syncthreads();
  for (int off = 128; off > 0; off >>= 1) {
    if (tid < off) { r1[tid] += r1[tid + off]; r2[tid] += r2[tid + off]; }
    __syncthreads();
  }
  const float mean = r1[0] * (1.0f / DMODEL);
  const float var  = r2[0] * (1.0f / DMODEL) - mean * mean;
  const float rstd = rsqrtf(var + 1e-5f);
  __bf16* o = out + (size_t)tok * DMODEL;
  o[tid]       = f2bf((v0 - mean) * rstd * w[tid]       + b[tid]);
  o[tid + 256] = f2bf((v1 - mean) * rstd * w[tid + 256] + b[tid + 256]);
  o[tid + 512] = f2bf((v2 - mean) * rstd * w[tid + 512] + b[tid + 512]);
}

// ---------------------------------------------------------------------------
// bf16 WMMA GEMM: out = A(bf16, MxK) @ W(f32->bf16, KxN) + bias, with modes:
//   0: store bf16                 (QKV projections)
//   1: f32 residual add in place  (attn proj, fc2)
//   2: GELU then store bf16       (fc1)
// Block tile 128x128x32, 8 waves, wave tile 32x64 (2x4 wmma tiles).
// A tile staged via GLOBAL_LOAD_ASYNC_TO_LDS_B128 (ASYNCcnt);
// W tile converted f32->bf16 with native cvt, packed ds_store_b32.
// ---------------------------------------------------------------------------
__global__ __launch_bounds__(256) void gemm_kernel(
    const __bf16* __restrict__ A, const float* __restrict__ W,
    const float* __restrict__ bias, void* __restrict__ outp,
    int M, int N, int K, int mode) {
  __shared__ __attribute__((aligned(16))) __bf16 As[128 * 40]; // [m][k], stride 40
  __shared__ __attribute__((aligned(16))) __bf16 Bs[128 * 40]; // [n][k], stride 40

  const int tid    = threadIdx.x;
  const int lane   = tid & 31;
  const int wid    = tid >> 5;
  const int lane_n = lane & 15;
  const int hw     = lane >> 4;      // half-wave
  const int kb     = hw * 8;         // A-fragment K interleave base
  const int m0     = blockIdx.y * 128;
  const int n0     = blockIdx.x * 128;
  const int wave_m = (wid >> 1) * 32;
  const int wave_n = (wid & 1) * 64;

  v8f acc[2][4];
#pragma unroll
  for (int mi = 0; mi < 2; ++mi)
#pragma unroll
    for (int ni = 0; ni < 4; ++ni)
#pragma unroll
      for (int r = 0; r < 8; ++r) acc[mi][ni][r] = 0.0f;

  const int arow = tid >> 1, ac0 = (tid & 1) * 16;     // A-tile loader mapping
  const int kpr  = (tid >> 4) * 2, wn0 = (tid & 15) * 8; // W-tile loader mapping
  const unsigned a_lds = lds_off(As + arow * 40 + ac0);  // 16B aligned
  const int ktiles = K >> 5;

  for (int kt = 0; kt < ktiles; ++kt) {
    const int k0 = kt << 5;
    // ---- stage A tile: async copy global (bf16) -> LDS, 2 x 16B per thread
    {
      const __bf16* ag = A + (size_t)(m0 + arow) * K + k0 + ac0;
      asm volatile(
          "global_load_async_to_lds_b128 %0, %1, off\n\t"
          "global_load_async_to_lds_b128 %0, %1, off offset:16"
          :
          : "v"(a_lds), "v"((unsigned long long)(size_t)ag)
          : "memory");
    }
    // ---- stage W tile: f32 -> bf16, transposed to N-major, packed b32 stores
    {
      const float* wg0 = W + (size_t)(k0 + kpr) * N + n0 + wn0;
      const float* wg1 = wg0 + N;
#pragma unroll
      for (int i = 0; i < 8; ++i) {
        v2bf pv;
        pv[0] = f2bf(wg0[i]);   // (n, kpr)
        pv[1] = f2bf(wg1[i]);   // (n, kpr+1)
        *(v2bf*)(Bs + (wn0 + i) * 40 + kpr) = pv;
      }
      if (kt + 1 < ktiles) __builtin_prefetch(wg0 + (size_t)32 * N, 0, 0);
    }
    asm volatile("s_wait_asynccnt 0x0" ::: "memory");   // drain async A copies
    __syncthreads();

    v16bf af[2], bfr[4];
#pragma unroll
    for (int mi = 0; mi < 2; ++mi) {
      const __bf16* p = As + (wave_m + mi * 16 + lane_n) * 40 + kb;
      af[mi] = pack16(*(const v8bf*)p, *(const v8bf*)(p + 16));
    }
#pragma unroll
    for (int ni = 0; ni < 4; ++ni) {
      const __bf16* p = Bs + (wave_n + ni * 16 + lane_n) * 40 + kb;
      bfr[ni] = pack16(*(const v8bf*)p, *(const v8bf*)(p + 16));
    }
#pragma unroll
    for (int mi = 0; mi < 2; ++mi)
#pragma unroll
      for (int ni = 0; ni < 4; ++ni)
        acc[mi][ni] = __builtin_amdgcn_wmma_f32_16x16x32_bf16(
            false, af[mi], false, bfr[ni], (short)0, acc[mi][ni], false, false);
    __syncthreads();
  }

  // epilogue
#pragma unroll
  for (int mi = 0; mi < 2; ++mi) {
#pragma unroll
    for (int ni = 0; ni < 4; ++ni) {
      const int col   = n0 + wave_n + ni * 16 + lane_n;
      const float bv  = bias[col];
      const int rbase = m0 + wave_m + mi * 16 + hw * 8;
#pragma unroll
      for (int r = 0; r < 8; ++r) {
        const size_t idx = (size_t)(rbase + r) * N + col;
        const float val = acc[mi][ni][r] + bv;
        if (mode == 0) {
          ((__bf16*)outp)[idx] = f2bf(val);
        } else if (mode == 1) {
          float* o = (float*)outp;
          o[idx] = o[idx] + val;
        } else {
          const float g = 0.5f * val * (1.0f + erff(val * 0.70710678f));
          ((__bf16*)outp)[idx] = f2bf(g);
        }
      }
    }
  }
}

// ---------------------------------------------------------------------------
// Flash attention, one wave per 16-query tile per (b,h).
// 3x3-tiled causal mask: allowed iff (i % 512) >= (j % 512).
// ---------------------------------------------------------------------------
__global__ __launch_bounds__(32) void attn_kernel(
    const __bf16* __restrict__ qm, const __bf16* __restrict__ km,
    const __bf16* __restrict__ vm, __bf16* __restrict__ ym) {
  const int lane   = threadIdx.x;
  const int lane_n = lane & 15;
  const int hw     = lane >> 4;
  const int kb     = hw * 8;
  const int q0     = blockIdx.x * 16;
  const int b      = blockIdx.y / NHEAD;
  const int h      = blockIdx.y % NHEAD;
  const int p0     = q0 & (SEGLEN - 1);
  const float scale = 0.125f;          // 1/sqrt(64)
  const int rowstride = NHEAD * HDIM;  // 768 elements per token

  __shared__ __attribute__((aligned(16))) __bf16 Pt[16 * 40];

  // Q fragments (A operand), rows q0+lane_n, K-dims 0..63 as two k=32 chunks
  const __bf16* qrow = qm + ((size_t)((b * SEQ + q0 + lane_n) * NHEAD + h)) * HDIM;
  v16bf aq[2];
#pragma unroll
  for (int f = 0; f < 2; ++f)
    aq[f] = *(const v16bf*)(qrow + f * 32 + hw * 16);

  float mrun[8], lrun[8];
  v8f oacc[4];
#pragma unroll
  for (int r = 0; r < 8; ++r) { mrun[r] = -1e30f; lrun[r] = 0.0f; }
#pragma unroll
  for (int n = 0; n < 4; ++n)
#pragma unroll
    for (int r = 0; r < 8; ++r) oacc[n][r] = 0.0f;

  for (int seg = 0; seg < 3; ++seg) {
    const int j0s  = seg * SEGLEN;
    const int jend = p0 + 16;               // keys with (j%512) <= p0+15 can pass
    for (int j = 0; j < jend; j += 32) {
      float s0[8], s1[8];
#pragma unroll
      for (int st = 0; st < 2; ++st) {
        // B operand of S = Q*K^T: lane holds column (key) lane_n, K-dims hw*16..+15
        const __bf16* krow =
            km + ((size_t)((b * SEQ + j0s + j + st * 16 + lane_n) * NHEAD + h)) * HDIM;
        v16bf bk0 = *(const v16bf*)(krow + hw * 16);
        v16bf bk1 = *(const v16bf*)(krow + 32 + hw * 16);
        v8f c;
#pragma unroll
        for (int r = 0; r < 8; ++r) c[r] = 0.0f;
        c = __builtin_amdgcn_wmma_f32_16x16x32_bf16(false, aq[0], false, bk0, (short)0, c, false, false);
        c = __builtin_amdgcn_wmma_f32_16x16x32_bf16(false, aq[1], false, bk1, (short)0, c, false, false);
        const int jl = j + st * 16 + lane_n;
#pragma unroll
        for (int r = 0; r < 8; ++r) {
          const int prow = p0 + r + hw * 8;
          const float sv = c[r] * scale;
          const float msk = (jl <= prow) ? sv : -1e30f;
          if (st == 0) s0[r] = msk; else s1[r] = msk;
        }
      }
      // online softmax update (row reductions across the 16-lane half groups)
#pragma unroll
      for (int r = 0; r < 8; ++r) {
        float mx = fmaxf(s0[r], s1[r]);
#pragma unroll
        for (int d = 8; d >= 1; d >>= 1) mx = fmaxf(mx, __shfl_xor(mx, d, 16));
        const float mnew = fmaxf(mrun[r], mx);
        const float e0 = __expf(s0[r] - mnew);
        const float e1 = __expf(s1[r] - mnew);
        s0[r] = e0; s1[r] = e1;
        float rs = e0 + e1;
#pragma unroll
        for (int d = 8; d >= 1; d >>= 1) rs += __shfl_xor(rs, d, 16);
        const float alpha = __expf(mrun[r] - mnew);
        lrun[r] = lrun[r] * alpha + rs;
        mrun[r] = mnew;
#pragma unroll
        for (int n = 0; n < 4; ++n) oacc[n][r] *= alpha;
      }
      // C-layout -> A-layout conversion of P through LDS
      __syncthreads();
#pragma unroll
      for (int r = 0; r < 8; ++r) {
        Pt[(r + hw * 8) * 40 + lane_n]      = f2bf(s0[r]);
        Pt[(r + hw * 8) * 40 + 16 + lane_n] = f2bf(s1[r]);
      }
      __syncthreads();
      const __bf16* pr = Pt + lane_n * 40 + hw * 16;
      const v16bf ap = pack16(*(const v8bf*)pr, *(const v8bf*)(pr + 8));
      // O += P @ V  (B operand: lane holds dim column lane_n, keys hw*16..+15)
#pragma unroll
      for (int n = 0; n < 4; ++n) {
        const __bf16* vbase =
            vm + ((size_t)((b * SEQ + j0s + j) * NHEAD + h)) * HDIM + n * 16 + lane_n;
        v16bf bvf;
#pragma unroll
        for (int i = 0; i < 16; ++i)
          bvf[i] = vbase[(size_t)(hw * 16 + i) * rowstride];
        oacc[n] = __builtin_amdgcn_wmma_f32_16x16x32_bf16(
            false, ap, false, bvf, (short)0, oacc[n], false, false);
      }
    }
  }
  // normalize and store bf16
#pragma unroll
  for (int n = 0; n < 4; ++n)
#pragma unroll
    for (int r = 0; r < 8; ++r) {
      const int row = q0 + r + hw * 8;
      const float val = oacc[n][r] / lrun[r];
      ym[((size_t)((b * SEQ + row) * NHEAD + h)) * HDIM + n * 16 + lane_n] = f2bf(val);
    }
}

// ---------------------------------------------------------------------------
// Host orchestration
// ---------------------------------------------------------------------------
extern "C" void kernel_launch(void* const* d_in, const int* in_sizes, int n_in,
                              void* d_out, int out_size, void* d_ws, size_t ws_size,
                              hipStream_t stream) {
  (void)in_sizes; (void)n_in; (void)out_size;
  const int*   idx_up   = (const int*)d_in[0];
  const int*   idx_down = (const int*)d_in[1];
  const float* cond     = (const float*)d_in[2];
  const float* teu      = (const float*)d_in[3];
  const float* ted      = (const float*)d_in[4];
  const float* pos      = (const float*)d_in[5];
  const float* cw       = (const float*)d_in[6];
  const float* cbias    = (const float*)d_in[7];
  const float* ln1w     = (const float*)d_in[8];
  const float* ln1b     = (const float*)d_in[9];
  const float* ln2w     = (const float*)d_in[10];
  const float* ln2b     = (const float*)d_in[11];
  const float* Wq = (const float*)d_in[12]; const float* bq = (const float*)d_in[13];
  const float* Wk = (const float*)d_in[14]; const float* bk = (const float*)d_in[15];
  const float* Wv = (const float*)d_in[16]; const float* bv = (const float*)d_in[17];
  const float* Wp = (const float*)d_in[18]; const float* bp = (const float*)d_in[19];
  const float* W1 = (const float*)d_in[20]; const float* b1 = (const float*)d_in[21];
  const float* W2 = (const float*)d_in[22]; const float* b2 = (const float*)d_in[23];

  // workspace layout (~198 MB)
  const size_t XB = (size_t)MTOK * DMODEL * sizeof(float);   // 37,748,736
  const size_t HB = (size_t)MTOK * DMODEL * sizeof(__bf16);  // 18,874,368
  const size_t GB = (size_t)MTOK * DFF    * sizeof(__bf16);  // 75,497,472
  if (ws_size < XB + 5 * HB + GB) return;
  char* ws = (char*)d_ws;
  float*  x    = (float*)(ws);
  __bf16* hb   = (__bf16*)(ws + XB);
  __bf16* qbuf = (__bf16*)(ws + XB + 1 * HB);
  __bf16* kbuf = (__bf16*)(ws + XB + 2 * HB);
  __bf16* vbuf = (__bf16*)(ws + XB + 3 * HB);
  __bf16* ybuf = (__bf16*)(ws + XB + 4 * HB);
  __bf16* gbuf = (__bf16*)(ws + XB + 5 * HB);

  dim3 g768(DMODEL / 128, MTOK / 128);   // (6, 96)
  dim3 g3072(DFF / 128, MTOK / 128);     // (24, 96)
  dim3 gattn(SEQ / 16, BATCH * NHEAD);   // (96, 96)

  embed_kernel<<<MTOK, 256, 0, stream>>>(idx_up, idx_down, cond, teu, ted, pos, cw, cbias, x);

  for (int l = 0; l < LAYERS; ++l) {
    const size_t wdd = (size_t)l * DMODEL * DMODEL;
    const size_t wdf = (size_t)l * DMODEL * DFF;

    ln_kernel<<<MTOK, 256, 0, stream>>>(x, ln1w + l * DMODEL, ln1b + l * DMODEL, hb);
    gemm_kernel<<<g768, 256, 0, stream>>>(hb, Wq + wdd, bq + l * DMODEL, qbuf, MTOK, DMODEL, DMODEL, 0);
    gemm_kernel<<<g768, 256, 0, stream>>>(hb, Wk + wdd, bk + l * DMODEL, kbuf, MTOK, DMODEL, DMODEL, 0);
    gemm_kernel<<<g768, 256, 0, stream>>>(hb, Wv + wdd, bv + l * DMODEL, vbuf, MTOK, DMODEL, DMODEL, 0);
    attn_kernel<<<gattn, 32, 0, stream>>>(qbuf, kbuf, vbuf, ybuf);
    gemm_kernel<<<g768, 256, 0, stream>>>(ybuf, Wp + wdd, bp + l * DMODEL, x, MTOK, DMODEL, DMODEL, 1);

    ln_kernel<<<MTOK, 256, 0, stream>>>(x, ln2w + l * DMODEL, ln2b + l * DMODEL, hb);
    gemm_kernel<<<g3072, 256, 0, stream>>>(hb, W1 + wdf, b1 + l * DFF, gbuf, MTOK, DFF, DMODEL, 2);
    gemm_kernel<<<g768, 256, 0, stream>>>(gbuf, W2 + wdf, b2 + l * DMODEL, x, MTOK, DMODEL, DFF, 1);
  }

  hipMemcpyAsync(d_out, x, XB, hipMemcpyDeviceToDevice, stream);
}
